// LambdaLayer_13417477833563
// MI455X (gfx1250) — matrix operand order repaired and checked by reading, not compiled, not run on hardware
//
#include <hip/hip_runtime.h>
#include <hip/hip_bf16.h>

// ---------------- problem constants ----------------
#define NB   16      // batch
#define NC   256     // input channels
#define NM   4096    // H*W
#define NK   16      // k
#define NU   4       // u
#define NV   256     // v
#define NUK  64      // u*k
#define NUV  1024    // u*v
#define NOUT 1104    // 16 + 64 + 1024 stacked output channels
#define BNEPS 1e-5f

typedef __attribute__((ext_vector_type(16))) __bf16 v16bf;
typedef __attribute__((ext_vector_type(8)))  float  v8f;
typedef __attribute__((ext_vector_type(2)))  float  v2f;

union Frag16 { v16bf v; uint4 q[2]; };

// ---------------- workspace layout (bytes) ----------------
// Wbf   : 1104*256 bf16            =    565,248
// Qbuf  : 16*16*4096 f32           =  4,194,304   [b][k][m]
// Kbuf  : 16*64*4096 f32           = 16,777,216   [b][uk][m]
// Vt    : 16*4096*1024 f32         = 268,435,456  [b][m][uv]  (transposed)
// ctxp  : 16 slices * 16*16*256 f32=  4,194,304
// ctx   : 16*16*256 f32            =    262,144
// qsb   : 32 f32 (scale[16],bias[16])
// vsb   : 2048 f32 (scale[1024],bias[1024])
// ksm   : 2048 f32 (rowmax[1024],rowinv[1024])
static constexpr size_t WBF_OFF  = 0;
static constexpr size_t QBUF_OFF = 565248ull;
static constexpr size_t KBUF_OFF = QBUF_OFF + 4194304ull;
static constexpr size_t VT_OFF   = KBUF_OFF + 16777216ull;
static constexpr size_t CTXP_OFF = VT_OFF   + 268435456ull;
static constexpr size_t CTX_OFF  = CTXP_OFF + 4194304ull;
static constexpr size_t QSB_OFF  = CTX_OFF  + 262144ull;
static constexpr size_t VSB_OFF  = QSB_OFF  + 128ull;
static constexpr size_t KSM_OFF  = VSB_OFF  + 8192ull;

// ---------------- 0: weights f32 -> bf16 (stacked [1104][256]) ----------------
__global__ __launch_bounds__(256) void convert_weights_kernel(
    const float* __restrict__ Wq, const float* __restrict__ Wk,
    const float* __restrict__ Wv, __bf16* __restrict__ Wbf)
{
  int i = blockIdx.x * 256 + threadIdx.x;
  if (i >= NOUT * NC) return;
  int o = i / NC, c = i % NC;
  float w;
  if (o < NK)            w = Wq[o * NC + c];
  else if (o < NK + NUK) w = Wk[(o - NK) * NC + c];
  else                   w = Wv[(o - 80) * NC + c];
  Wbf[i] = (__bf16)w;
}

// ---------------- 1: fused QKV projection (bf16 WMMA, f32 accumulate) -------
// One block per (batch, 16-pixel m-tile). x tile staged in LDS as bf16
// [16 m][256 c] (c-contiguous). 8 waves sweep the 69 output-channel tiles.
//   Q/K tiles: D = W[16o x 32c] * x[32c x 16m]   -> store [b][o][m]
//   V   tiles: D = xT[16m x 32c] * WT[32c x 16o] -> store [b][m][uv] (Vt)
__global__ __launch_bounds__(256) void proj_kernel(
    const float* __restrict__ x, const __bf16* __restrict__ Wbf,
    const float* __restrict__ bq, const float* __restrict__ bk,
    const float* __restrict__ bv,
    float* __restrict__ Qbuf, float* __restrict__ Kbuf, float* __restrict__ Vt)
{
  __shared__ __bf16 Xs[16][NC + 8];   // 16 x 264 bf16, rows 528B (16B aligned)
  const int b  = blockIdx.x >> 8;
  const int mb = (blockIdx.x & 255) * 16;
  const int t  = threadIdx.x;
  {
    const int mloc = t & 15;
    const int cb   = t >> 4;
    const float* xb = x + ((size_t)b * NC) * NM + mb + mloc;
    #pragma unroll
    for (int p = 0; p < 16; ++p) {
      int c = cb + p * 16;
      Xs[mloc][c] = (__bf16)xb[(size_t)c * NM];
    }
  }
  __syncthreads();
  const int lane = t & 31, wave = t >> 5;
  const int l15 = lane & 15, h = lane >> 4;

  for (int ot = wave; ot < 69; ot += 8) {
    const int ob = ot * 16;
    v8f acc = {};
    if (ot < 5) {
      // ---- Q/K orientation: A = weight rows, B = x from LDS ----
      const __bf16* wrow = Wbf + (size_t)(ob + l15) * NC;  // A row o = ob+l15
      const __bf16* xrow = &Xs[l15][0];                    // B col m = l15
      #pragma unroll
      for (int c0 = 0; c0 < NC; c0 += 32) {
        Frag16 Af, Bf;
        Af.q[0] = *(const uint4*)(wrow + c0 + h * 8);        // K = c0+hb..+7
        Af.q[1] = *(const uint4*)(wrow + c0 + h * 8 + 16);   // K = c0+hb+16..+23
        Bf.q[0] = *(const uint4*)(xrow + c0 + h * 8);        // K = c0+hb..+7
        Bf.q[1] = *(const uint4*)(xrow + c0 + h * 8 + 16);   // K = c0+hb+16..+23
        acc = __builtin_amdgcn_wmma_f32_16x16x32_bf16(
                false, Af.v, false, Bf.v, (short)0, acc, false, false);
      }
      const int m = mb + l15;
      if (ot == 0) {
        #pragma unroll
        for (int r = 0; r < 8; ++r) {
          int o = r + h * 8;
          Qbuf[((size_t)b * NK + o) * NM + m] = acc[r] + bq[o];
        }
      } else {
        #pragma unroll
        for (int r = 0; r < 8; ++r) {
          int o = ob - NK + r + h * 8;
          Kbuf[((size_t)b * NUK + o) * NM + m] = acc[r] + bk[o];
        }
      }
    } else {
      // ---- V orientation: A = xT from LDS, B = WT rows; D rows = m ----
      const int uv = ob - 80 + l15;
      const __bf16* wrow = Wbf + (size_t)(ob + l15) * NC;  // B col o = ob+l15
      const __bf16* xrow = &Xs[l15][0];                    // A row m = l15
      #pragma unroll
      for (int c0 = 0; c0 < NC; c0 += 32) {
        Frag16 Af, Bf;
        Af.q[0] = *(const uint4*)(xrow + c0 + h * 8);
        Af.q[1] = *(const uint4*)(xrow + c0 + h * 8 + 16);
        Bf.q[0] = *(const uint4*)(wrow + c0 + h * 8);
        Bf.q[1] = *(const uint4*)(wrow + c0 + h * 8 + 16);
        acc = __builtin_amdgcn_wmma_f32_16x16x32_bf16(
                false, Af.v, false, Bf.v, (short)0, acc, false, false);
      }
      const float bias = bv[uv];
      #pragma unroll
      for (int r = 0; r < 8; ++r) {
        int m = mb + r + h * 8;
        Vt[((size_t)b * NM + m) * NUV + uv] = acc[r] + bias;   // coalesced in uv
      }
    }
  }
}

// ---------------- 2: BN stats for Q -> qscale/qbias (folded affine) ---------
__global__ __launch_bounds__(256) void q_stats_kernel(
    const float* __restrict__ Qbuf, const float* __restrict__ gq,
    const float* __restrict__ btq, float* __restrict__ qsb)
{
  const int ch = blockIdx.x;   // 0..15
  __shared__ float s1[256], s2[256];
  float s = 0.f, sq = 0.f;
  for (int i = threadIdx.x; i < NB * NM; i += 256) {
    int b = i >> 12, m = i & 4095;
    float v = Qbuf[((size_t)b * NK + ch) * NM + m];
    s += v; sq += v * v;
  }
  s1[threadIdx.x] = s; s2[threadIdx.x] = sq;
  __syncthreads();
  for (int st = 128; st > 0; st >>= 1) {
    if (threadIdx.x < st) {
      s1[threadIdx.x] += s1[threadIdx.x + st];
      s2[threadIdx.x] += s2[threadIdx.x + st];
    }
    __syncthreads();
  }
  if (threadIdx.x == 0) {
    const float n = (float)(NB * NM);
    float mean = s1[0] / n;
    float var  = s2[0] / n - mean * mean;
    float sc = gq[ch] * rsqrtf(var + BNEPS);
    qsb[ch]      = sc;
    qsb[NK + ch] = btq[ch] - mean * sc;
  }
}

// ---------------- 3: BN stats for V (one thread per channel, coalesced) -----
__global__ __launch_bounds__(256) void v_stats_kernel(
    const float* __restrict__ Vt, const float* __restrict__ gv,
    const float* __restrict__ btv, float* __restrict__ vsb)
{
  const int ch = blockIdx.x * 256 + threadIdx.x;   // 0..1023
  float s = 0.f, sq = 0.f;
  #pragma unroll 4
  for (int bm = 0; bm < NB * NM; ++bm) {
    float v = Vt[(size_t)bm * NUV + ch];
    s += v; sq += v * v;
  }
  const float n = (float)(NB * NM);
  float mean = s / n;
  float var  = sq / n - mean * mean;
  float sc = gv[ch] * rsqrtf(var + BNEPS);
  vsb[ch]       = sc;
  vsb[NUV + ch] = btv[ch] - mean * sc;
}

// ---------------- 4: softmax row stats for K (max, 1/sumexp) ----------------
__global__ __launch_bounds__(256) void k_softmax_stats_kernel(
    const float* __restrict__ Kbuf, float* __restrict__ ksm)
{
  const int row = blockIdx.x;               // b*64 + uk, 0..1023
  const float* kr = Kbuf + (size_t)row * NM;
  __shared__ float sh[256];
  float mx = -3.4e38f;
  for (int i = threadIdx.x; i < NM; i += 256) mx = fmaxf(mx, kr[i]);
  sh[threadIdx.x] = mx; __syncthreads();
  for (int st = 128; st > 0; st >>= 1) {
    if (threadIdx.x < st) sh[threadIdx.x] = fmaxf(sh[threadIdx.x], sh[threadIdx.x + st]);
    __syncthreads();
  }
  mx = sh[0]; __syncthreads();
  float s = 0.f;
  for (int i = threadIdx.x; i < NM; i += 256) s += __expf(kr[i] - mx);
  sh[threadIdx.x] = s; __syncthreads();
  for (int st = 128; st > 0; st >>= 1) {
    if (threadIdx.x < st) sh[threadIdx.x] += sh[threadIdx.x + st];
    __syncthreads();
  }
  if (threadIdx.x == 0) {
    ksm[row]            = mx;
    ksm[NB * NUK + row] = 1.0f / sh[0];
  }
}

// ---------------- 5: context = softmax(K) @ Vn^T (f32 WMMA, folded BN) -----
// block = (b, m-slice of 256). Wave w owns v-tiles {2w,2w+1}; softmax and
// V-BN are applied while building the A/B fragments.
__global__ __launch_bounds__(256) void context_kernel(
    const float* __restrict__ Kbuf, const float* __restrict__ Vt,
    const float* __restrict__ ksm, const float* __restrict__ vsb,
    float* __restrict__ ctxp)
{
  const int b = blockIdx.x >> 4;
  const int slice = blockIdx.x & 15;
  const int t = threadIdx.x, lane = t & 31, wave = t >> 5;
  const int l15 = lane & 15, h = lane >> 4;
  const int vt0 = wave * 2, vt1 = wave * 2 + 1;
  v8f acc0 = {}, acc1 = {};
  const float* vtb = Vt + ((size_t)b * NM) * NUV;
  const int mlo = slice * 256;

  for (int u = 0; u < NU; ++u) {
    const int krow = b * NUK + u * NK + l15;          // A row k = l15
    const float mx  = ksm[krow];
    const float inv = ksm[NB * NUK + krow];
    const float* kr = Kbuf + (size_t)krow * NM;
    const int uv0 = u * NV + vt0 * 16 + l15;          // B col v (tile 0)
    const int uv1 = u * NV + vt1 * 16 + l15;          // B col v (tile 1)
    const float vs0 = vsb[uv0], vo0 = vsb[NUV + uv0];
    const float vs1 = vsb[uv1], vo1 = vsb[NUV + uv1];
    for (int m0 = mlo; m0 < mlo + 256; m0 += 4) {
      const int mc = m0 + h * 2;                      // K = mc, mc+1 per lane
      if (m0 + 32 < NM)
        __builtin_prefetch(vtb + (size_t)(mc + 32) * NUV + uv0, 0, 0);
      v2f a, b0, b1;
      a.x = __expf(kr[mc]     - mx) * inv;
      a.y = __expf(kr[mc + 1] - mx) * inv;
      const float* vp = vtb + (size_t)mc * NUV;
      b0.x = vp[uv0]       * vs0 + vo0;
      b0.y = vp[NUV + uv0] * vs0 + vo0;
      b1.x = vp[uv1]       * vs1 + vo1;
      b1.y = vp[NUV + uv1] * vs1 + vo1;
      acc0 = __builtin_amdgcn_wmma_f32_16x16x4_f32(
               false, a, false, b0, (short)0, acc0, false, false);
      acc1 = __builtin_amdgcn_wmma_f32_16x16x4_f32(
               false, a, false, b1, (short)0, acc1, false, false);
    }
  }
  float* cp = ctxp + (size_t)slice * (NB * NK * NV) + (size_t)b * (NK * NV);
  #pragma unroll
  for (int r = 0; r < 8; ++r) {
    int k = r + h * 8;
    cp[(size_t)k * NV + vt0 * 16 + l15] = acc0[r];
    cp[(size_t)k * NV + vt1 * 16 + l15] = acc1[r];
  }
}

// ---------------- 6: deterministic reduce of the 16 context partials --------
__global__ __launch_bounds__(256) void ctx_reduce_kernel(
    const float* __restrict__ ctxp, float* __restrict__ ctx)
{
  int i = blockIdx.x * 256 + threadIdx.x;
  if (i >= NB * NK * NV) return;
  float s = 0.f;
  #pragma unroll
  for (int p = 0; p < 16; ++p) s += ctxp[(size_t)p * (NB * NK * NV) + i];
  ctx[i] = s;
}

// ---------------- 7: y = ctx^T @ Qn (f32 WMMA, Q-BN folded into B load) ----
__global__ __launch_bounds__(256) void final_kernel(
    const float* __restrict__ Qbuf, const float* __restrict__ ctx,
    const float* __restrict__ qsb, float* __restrict__ y)
{
  const int blk = blockIdx.x;          // (b*16 + vt)*4 + mchunk, 0..1023
  const int mchunk = blk & 3;
  const int vt = (blk >> 2) & 15;
  const int b = blk >> 6;
  const int t = threadIdx.x, lane = t & 31, wave = t >> 5;
  const int l15 = lane & 15, h = lane >> 4;
  const float* cb = ctx + (size_t)b * (NK * NV);
  const int v = vt * 16 + l15;         // A row (v) for this lane
  v2f A[4];
  float qs0[4], qo0[4], qs1[4], qo1[4];
  #pragma unroll
  for (int j = 0; j < 4; ++j) {
    int kc = j * 4 + h * 2;            // K indices kc, kc+1 per lane
    A[j].x = cb[(size_t)kc * NV + v];
    A[j].y = cb[(size_t)(kc + 1) * NV + v];
    qs0[j] = qsb[kc];      qo0[j] = qsb[NK + kc];
    qs1[j] = qsb[kc + 1];  qo1[j] = qsb[NK + kc + 1];
  }
  const float* qb = Qbuf + (size_t)b * NK * NM;
  float* yb = y + (size_t)(b * NV + vt * 16) * NM;
  for (int mt = mchunk * 64 + wave; mt < (mchunk + 1) * 64; mt += 8) {
    const int m = mt * 16 + l15;
    v8f acc = {};
    #pragma unroll
    for (int j = 0; j < 4; ++j) {
      int kc = j * 4 + h * 2;
      v2f Bf;
      Bf.x = qb[(size_t)kc * NM + m]       * qs0[j] + qo0[j];
      Bf.y = qb[(size_t)(kc + 1) * NM + m] * qs1[j] + qo1[j];
      acc = __builtin_amdgcn_wmma_f32_16x16x4_f32(
              false, A[j], false, Bf, (short)0, acc, false, false);
    }
    #pragma unroll
    for (int r = 0; r < 8; ++r)
      yb[(size_t)(r + h * 8) * NM + m] = acc[r];     // coalesced along m
  }
}

// ---------------- launch ----------------
extern "C" void kernel_launch(void* const* d_in, const int* in_sizes, int n_in,
                              void* d_out, int out_size, void* d_ws, size_t ws_size,
                              hipStream_t stream) {
  (void)in_sizes; (void)n_in; (void)out_size; (void)ws_size;
  const float* x   = (const float*)d_in[0];
  const float* Wq  = (const float*)d_in[1];
  const float* bq  = (const float*)d_in[2];
  const float* Wk  = (const float*)d_in[3];
  const float* bk  = (const float*)d_in[4];
  const float* Wv  = (const float*)d_in[5];
  const float* bv  = (const float*)d_in[6];
  const float* gq  = (const float*)d_in[7];
  const float* btq = (const float*)d_in[8];
  const float* gv  = (const float*)d_in[9];
  const float* btv = (const float*)d_in[10];

  char* ws = (char*)d_ws;
  __bf16* Wbf  = (__bf16*)(ws + WBF_OFF);
  float*  Qbuf = (float*)(ws + QBUF_OFF);
  float*  Kbuf = (float*)(ws + KBUF_OFF);
  float*  Vt   = (float*)(ws + VT_OFF);
  float*  ctxp = (float*)(ws + CTXP_OFF);
  float*  ctxb = (float*)(ws + CTX_OFF);
  float*  qsb  = (float*)(ws + QSB_OFF);
  float*  vsb  = (float*)(ws + VSB_OFF);
  float*  ksm  = (float*)(ws + KSM_OFF);

  convert_weights_kernel<<<(NOUT * NC + 255) / 256, 256, 0, stream>>>(Wq, Wk, Wv, Wbf);
  proj_kernel<<<NB * (NM / 16), 256, 0, stream>>>(x, Wbf, bq, bk, bv, Qbuf, Kbuf, Vt);
  q_stats_kernel<<<NK, 256, 0, stream>>>(Qbuf, gq, btq, qsb);
  v_stats_kernel<<<NUV / 256, 256, 0, stream>>>(Vt, gv, btv, vsb);
  k_softmax_stats_kernel<<<NB * NUK, 256, 0, stream>>>(Kbuf, ksm);
  context_kernel<<<NB * 16, 256, 0, stream>>>(Kbuf, Vt, ksm, vsb, ctxp);
  ctx_reduce_kernel<<<(NB * NK * NV + 255) / 256, 256, 0, stream>>>(ctxp, ctxb);
  final_kernel<<<NB * 16 * 4, 256, 0, stream>>>(Qbuf, ctxb, qsb, (float*)d_out);
}
